// DeformableConv2dGuided_18416819765761
// MI455X (gfx1250) — compile-verified, matrix operand order checked by reading, not run on hardware
//
#include <hip/hip_runtime.h>
#include <hip/hip_bf16.h>
#include <math.h>

// Problem constants (reference: B=4, N=4, C_IN=64, C_OUT=64, H=W=192)
#define BB   4
#define NN_  4
#define CIN  64
#define COUT 64
#define HH   192
#define WW   192
#define HWSZ (HH * WW)          // 36864
#define CC   (NN_ * CIN)        // 256 packed input channels
#define KCONV (CC * 9)          // 2304 reduction length for the 3x3 conv
#define NCHUNK (KCONV / 32)     // 72 bf16 K=32 slabs

typedef __attribute__((ext_vector_type(16))) __bf16 v16bf;
typedef __attribute__((ext_vector_type(8)))  float  v8f;
typedef __attribute__((ext_vector_type(2)))  float  v2f;

#if defined(__has_builtin)
#if __has_builtin(__builtin_amdgcn_wmma_f32_16x16x4_f32)
#define HAS_WMMA_F32 1
#endif
#if __has_builtin(__builtin_amdgcn_perm)
#define HAS_PERM 1
#endif
#endif

union AB16 {
    v16bf v;
    unsigned int u[8];
    float4 f4[2];
};
union F2V {
    float2 f;
    v2f v;
};

// Accurate RNE (cold paths)
__device__ __forceinline__ unsigned int f32_to_bf16_rne(float f) {
    unsigned int u = __float_as_uint(f);
    return (u + 0x7FFFu + ((u >> 16) & 1u)) >> 16;
}
__device__ __forceinline__ unsigned int bf16pair_rne(float lo, float hi) {
    return (f32_to_bf16_rne(hi) << 16) | f32_to_bf16_rne(lo);
}
// Fast: round-half-up + one v_perm_b32 pack (hot path)
__device__ __forceinline__ unsigned int bf16pair_fast(float lo, float hi) {
    unsigned int a = __float_as_uint(lo) + 0x8000u;
    unsigned int b = __float_as_uint(hi) + 0x8000u;
#ifdef HAS_PERM
    return __builtin_amdgcn_perm(b, a, 0x07060302u);  // {b[31:16], a[31:16]}
#else
    return (b & 0xFFFF0000u) | (a >> 16);
#endif
}

// 16-bit A/B-matrix 16x32 half->K map (ISA 7.12.2):
// lanes 0-15: halves 0..7 -> K=0..7, halves 8..15 -> K=16..23 ; lanes 16-31: +8
__device__ __forceinline__ int kmap(int h, int laneHi) {
    return h + ((h >= 8) ? 8 : 0) + laneHi * 8;
}

// ---------------------------------------------------------------------------
// Kernel 0: pack conv weights (rows 0-7 = w_off, 8-11 = w_mod, 12-15 = zero)
// into bf16 A-matrix tiles: apack[chunk][lane][8 dwords].
// K ordering: k = tap*256 + ic  (tap = ky*3+kx over the 3x3 window).
// ---------------------------------------------------------------------------
__global__ __launch_bounds__(256) void pack_wts(
    const float* __restrict__ w_off, const float* __restrict__ w_mod,
    unsigned int* __restrict__ apack)
{
    int t = blockIdx.x * blockDim.x + threadIdx.x;
    if (t >= NCHUNK * 32) return;
    int chunk = t >> 5, lane = t & 31;
    int m = lane & 15, hi = (lane >> 4) & 1;
#pragma unroll
    for (int j = 0; j < 8; ++j) {
        float vv[2];
#pragma unroll
        for (int e = 0; e < 2; ++e) {
            int kg  = chunk * 32 + kmap(2 * j + e, hi);
            int tap = kg >> 8, ic = kg & 255;
            int ky = tap / 3, kx = tap % 3;
            float w;
            if (m < 8)       w = w_off[((m * CC + ic) * 3 + ky) * 3 + kx];
            else if (m < 12) w = w_mod[(((m - 8) * CC + ic) * 3 + ky) * 3 + kx];
            else             w = 0.0f;
            vv[e] = w;
        }
        apack[t * 8 + j] = bf16pair_rne(vv[0], vv[1]);
    }
}

// ---------------------------------------------------------------------------
// Kernel 0b (optional, ws-permitting): convert x to channel-pair-interleaved
// bf16: xpack[b*128 + cp][pixel] dword = {bf16(ch=2cp+1), bf16(ch=2cp)}.
// This matches the 16-bit WMMA operand layout so the conv B-operand becomes
// 8 raw b32 loads per K=32 step, zero conversion VALU.
// ---------------------------------------------------------------------------
__global__ __launch_bounds__(256) void cvt_x_bf16(
    const float* __restrict__ x, unsigned int* __restrict__ xp)
{
    int pix = blockIdx.x * 256 + threadIdx.x;       // 0 .. HWSZ-1  (144 blocks)
    int bcp = blockIdx.y;                           // 0 .. 511 (= B*128)
    size_t base = (size_t)(bcp * 2) * HWSZ + pix;
    xp[(size_t)bcp * HWSZ + pix] = bf16pair_rne(x[base], x[base + HWSZ]);
}

// ---------------------------------------------------------------------------
// Kernel 1: 3x3 conv -> offset (negated, +bias) and modulator (2*sigmoid).
// Implicit GEMM: M=16 (12 used outputs), N=16 pixels per wave, K=2304 via
// 72 x V_WMMA_F32_16X16X32_BF16 (fp32 accumulate).
// Loads are UNCONDITIONAL from clamped addresses; out-of-image taps are
// zeroed by selecting the packed dword (no exec-mask branches).
// ---------------------------------------------------------------------------
template <bool PACKED>
__global__ __launch_bounds__(256) void conv_off_mod(
    const float* __restrict__ x, const unsigned int* __restrict__ xpack,
    const unsigned int* __restrict__ apack,
    const float* __restrict__ b_off, const float* __restrict__ b_mod,
    float* __restrict__ out_offset, float* __restrict__ ws_mod)
{
    int wave = threadIdx.x >> 5, lane = threadIdx.x & 31;
    int tile = blockIdx.x * 8 + wave;                 // 0 .. B*H*12-1
    int b  = tile / (HH * 12);
    int r  = tile - b * (HH * 12);
    int y  = r / 12;
    int x0 = (r - y * 12) * 16;
    int n  = lane & 15, hi = lane >> 4;
    int px = x0 + n;

    const float*        xb  = x + (size_t)b * CC * HWSZ;
    const unsigned int* xpb = xpack + (size_t)b * (CC / 2) * HWSZ;
    v8f acc = {0.f, 0.f, 0.f, 0.f, 0.f, 0.f, 0.f, 0.f};

    for (int tap = 0; tap < 9; ++tap) {
        int dy = tap / 3 - 1, dx = tap % 3 - 1;
        int yy = y + dy, xx = px + dx;
        bool valid = (yy >= 0) & (yy < HH) & (xx >= 0) & (xx < WW);
        int yyc = yy < 0 ? 0 : (yy >= HH ? HH - 1 : yy);
        int xxc = xx < 0 ? 0 : (xx >= WW ? WW - 1 : xx);
        int saddr = yyc * WW + xxc;                    // always in-bounds
        for (int c8 = 0; c8 < 8; ++c8) {
            // A tile: 32B per lane, pre-packed bf16 layout
            AB16 a;
            const float4* ap =
                (const float4*)(apack + (((size_t)(tap * 8 + c8) * 32 + lane) * 8));
            a.f4[0] = ap[0];
            a.f4[1] = ap[1];
            // B tile: K=32 consecutive channels of this tap, N=16 pixels
            AB16 bm;
            if (PACKED) {
#pragma unroll
                for (int j = 0; j < 8; ++j) {
                    // dword j holds channels {2cp, 2cp+1}
                    int cp = c8 * 16 + ((j < 4) ? j : j + 4) + hi * 4;
                    unsigned int u = xpb[(size_t)cp * HWSZ + saddr];
                    bm.u[j] = valid ? u : 0u;
                }
            } else {
#pragma unroll
                for (int j = 0; j < 8; ++j) {
                    int ch0 = c8 * 32 + kmap(2 * j + 0, hi);
                    int ch1 = c8 * 32 + kmap(2 * j + 1, hi);
                    float t0 = xb[(size_t)ch0 * HWSZ + saddr];
                    float t1 = xb[(size_t)ch1 * HWSZ + saddr];
                    unsigned int p = bf16pair_fast(t0, t1);
                    bm.u[j] = valid ? p : 0u;
                }
            }
            acc = __builtin_amdgcn_wmma_f32_16x16x32_bf16(
                false, a.v, false, bm.v, (short)0, acc, false, false);
        }
    }

    // Epilogue: C/D 16x16 layout -> lane holds M = v + hi*8, col = lane&15
#pragma unroll
    for (int v = 0; v < 8; ++v) {
        int m = v + hi * 8;
        float d = acc[v];
        if (m < 8) {
            float off = -(d + b_off[m]);
            out_offset[(((size_t)b * 8 + m) * HH + y) * WW + px] = off;
        } else if (m < 12) {
            float tt  = d + b_mod[m - 8];
            float mod = 2.0f / (1.0f + expf(-tt));
            ws_mod[(((size_t)b * 4 + (m - 8)) * HH + y) * WW + px] = mod;
        }
    }
}

// ---------------------------------------------------------------------------
// Kernel 2: modulated bilinear gather + mean over N, then 1x1 conv (64x64)
// via V_WMMA_F32_16X16X4_F32. One wave per 32 pixels; sampled means staged
// in a padded LDS tile (64 x 33) for the K-major B-matrix reads.
// ---------------------------------------------------------------------------
__global__ __launch_bounds__(128) void gather_conv(
    const float* __restrict__ x, const float* __restrict__ offset,
    const float* __restrict__ ws_mod, const float* __restrict__ w_reg,
    float* __restrict__ out)
{
    __shared__ float lds[4 * 64 * 33];
    int wave = threadIdx.x >> 5, lane = threadIdx.x & 31;
    int tile = blockIdx.x * 4 + wave;                 // 0 .. B*H*6-1
    int b  = tile / (HH * 6);
    int r  = tile - b * (HH * 6);
    int y  = r / 6;
    int x0 = (r - y * 6) * 32;
    int xc = x0 + lane;

    // Per-lane: 16 taps (4 refs x 4 bilinear taps); weights fold modulator
    // and the 1/N mean; invalid taps get weight 0 (zero padding).
    size_t ofs[16];
    float  wt[16];
#pragma unroll
    for (int nn = 0; nn < 4; ++nn) {
        float oy  = offset[(((size_t)b * 8 + 2 * nn + 0) * HH + y) * WW + xc];
        float ox  = offset[(((size_t)b * 8 + 2 * nn + 1) * HH + y) * WW + xc];
        float mod = ws_mod[(((size_t)b * 4 + nn) * HH + y) * WW + xc] * 0.25f;
        float py  = (float)y + oy;
        float pxs = (float)xc + ox;
        float fy0 = floorf(py), fx0 = floorf(pxs);
        float wy = py - fy0, wx = pxs - fx0;
        int iy0 = (int)fy0, ix0 = (int)fx0;
        float tw[4] = {(1.f - wy) * (1.f - wx), (1.f - wy) * wx,
                       wy * (1.f - wx),         wy * wx};
        int ty[4] = {iy0, iy0, iy0 + 1, iy0 + 1};
        int tx[4] = {ix0, ix0 + 1, ix0, ix0 + 1};
        size_t nbase = (size_t)((b * 4 + nn) * CIN) * HWSZ;
#pragma unroll
        for (int t = 0; t < 4; ++t) {
            bool v = (ty[t] >= 0) & (ty[t] < HH) & (tx[t] >= 0) & (tx[t] < WW);
            int yy = v ? ty[t] : 0, xx = v ? tx[t] : 0;
            ofs[nn * 4 + t] = nbase + (size_t)(yy * WW + xx);
            wt[nn * 4 + t]  = v ? tw[t] * mod : 0.0f;
        }
    }

    // Gather phase: mean_sampled[c][pixel] -> LDS (stride 33: conflict-free)
    float* col = &lds[wave * 64 * 33];
    for (int c = 0; c < 64; ++c) {
        float s = 0.0f;
#pragma unroll
        for (int t = 0; t < 16; ++t)
            s += wt[t] * x[ofs[t] + (size_t)c * HWSZ];
        col[c * 33 + lane] = s;
    }
    __syncthreads();

    int m = lane & 15, hi = lane >> 4;
#ifdef HAS_WMMA_F32
    // 1x1 conv: out[o,p] = sum_c w_reg[o][c] * s[c][p]
    // A 16x4 f32: lane<16 -> K={0,1}, lane>=16 -> K={2,3}; B mirrored.
    for (int ot = 0; ot < 4; ++ot) {
        for (int nt = 0; nt < 2; ++nt) {
            v8f acc = {0.f, 0.f, 0.f, 0.f, 0.f, 0.f, 0.f, 0.f};
#pragma unroll
            for (int ks = 0; ks < 16; ++ks) {
                int kb = ks * 4 + hi * 2;              // even -> 8B aligned
                F2V a;
                a.f = *(const float2*)&w_reg[(ot * 16 + m) * 64 + kb];
                v2f bmat;
                bmat[0] = col[(kb + 0) * 33 + nt * 16 + m];
                bmat[1] = col[(kb + 1) * 33 + nt * 16 + m];
                acc = __builtin_amdgcn_wmma_f32_16x16x4_f32(
                    false, a.v, false, bmat, (short)0, acc, false, false);
            }
#pragma unroll
            for (int v = 0; v < 8; ++v) {
                int o = ot * 16 + v + hi * 8;
                out[(((size_t)b * COUT + o) * HH + y) * WW + x0 + nt * 16 + m] =
                    acc[v];
            }
        }
    }
#else
    // VALU fallback: each lane finishes its own pixel column.
    for (int o = 0; o < 64; ++o) {
        float a = 0.0f;
        for (int c = 0; c < 64; ++c)
            a += col[c * 33 + lane] * w_reg[o * 64 + c];
        out[(((size_t)b * COUT + o) * HH + y) * WW + xc] = a;
    }
#endif
}

// ---------------------------------------------------------------------------
// Workspace layout:
//   [0, 72KB)            bf16 A-pack (72 chunks * 1KB)
//   [128KB, +2.25MB)     modulator [B,4,H,W] fp32
//   [4MB, +72MB)         (optional) bf16 channel-pair-packed x
// d_out layout: out (B*64*H*W floats) then offset (B*8*H*W floats).
// Path choice depends only on ws_size (fixed per harness) -> deterministic.
// ---------------------------------------------------------------------------
extern "C" void kernel_launch(void* const* d_in, const int* in_sizes, int n_in,
                              void* d_out, int out_size, void* d_ws, size_t ws_size,
                              hipStream_t stream) {
    const float* x     = (const float*)d_in[0];
    const float* w_off = (const float*)d_in[1];
    const float* b_off = (const float*)d_in[2];
    const float* w_mod = (const float*)d_in[3];
    const float* b_mod = (const float*)d_in[4];
    const float* w_reg = (const float*)d_in[5];

    float* out        = (float*)d_out;
    float* out_offset = out + (size_t)BB * COUT * HH * WW;

    const size_t XPACK_OFF   = (size_t)4 << 20;
    const size_t XPACK_BYTES = (size_t)BB * (CC / 2) * HWSZ * sizeof(unsigned int);

    unsigned int* apack = (unsigned int*)d_ws;
    float* ws_mod = (float*)((char*)d_ws + (128u << 10));
    unsigned int* xpack = (unsigned int*)((char*)d_ws + XPACK_OFF);

    bool packed = ws_size >= XPACK_OFF + XPACK_BYTES;

    pack_wts<<<9, 256, 0, stream>>>(w_off, w_mod, apack);
    if (packed) {
        cvt_x_bf16<<<dim3(HWSZ / 256, BB * CC / 2), 256, 0, stream>>>(x, xpack);
        conv_off_mod<true><<<1152, 256, 0, stream>>>(x, xpack, apack, b_off,
                                                     b_mod, out_offset, ws_mod);
    } else {
        conv_off_mod<false><<<1152, 256, 0, stream>>>(x, xpack, apack, b_off,
                                                      b_mod, out_offset, ws_mod);
    }
    gather_conv<<<1152, 128, 0, stream>>>(x, out_offset, ws_mod, w_reg, out);
}
